// SoftDecisionTree_73555609911419
// MI455X (gfx1250) — matrix operand order repaired and checked by reading, not compiled, not run on hardware
//
#include <hip/hip_runtime.h>
#include <hip/hip_bf16.h>

// ---------------------------------------------------------------------------
// Soft decision tree, fully fused for gfx1250 (MI455X).
//   probs = sigmoid(beta * (x @ W.T + b))   -> bf16 WMMA GEMM, probs kept in LDS
//   loss  = tree-product reduction against softmax-leaf rewards (same kernel)
// HBM traffic: x 32MB + Wpack 2MB + out 32KB (no probs round-trip).
// ---------------------------------------------------------------------------

typedef __attribute__((ext_vector_type(16))) __bf16 v16bf;
typedef __attribute__((ext_vector_type(8)))  float  v8f;
typedef __attribute__((ext_vector_type(4)))  int    v4i;
typedef __attribute__((ext_vector_type(4)))  float  v4f;

#define BATCH   8192
#define DIM     1024
#define NNODES  1023
#define NPAD    1024      // node columns padded to 1024 (col 1023 = zeros)
#define NLEAVES 1024
#define NCLS    10

#define M_TILE  32        // rows per workgroup
#define XS_STR  1032      // LDS x-strip row stride, bf16 elems (bank-conflict-free b128)
#define PH_STR  1032      // LDS probs row stride, f16 elems
#define XS_BYTES (M_TILE * XS_STR * 2)              // 66048
#define LDS_BYTES (XS_BYTES + M_TILE * PH_STR * 2)  // 132096

// ws layout (bytes): [0,2MB) Wpack bf16 fragments; [2MB,+4KB) leaf_reward f32[1024]
#define WS_LRW_OFF (2u << 20)

static __device__ __forceinline__ unsigned short f2bf(float f) {
    unsigned u = __builtin_bit_cast(unsigned, f);
    u += 0x7fffu + ((u >> 16) & 1u);        // round-to-nearest-even
    return (unsigned short)(u >> 16);
}

// ---------------------------------------------------------------------------
// Kernel 1: pack W (f32 [node][dim]) into beta-folded bf16 WMMA B-fragments.
// Tile t = kt*64 + nt covers K rows [kt*32,+32), N cols [nt*16,+16); 1024 B:
// region h (0/1) of 512 B; lane l holds 16 B/region: N = l&15, khalf = l>>4,
// dword dv = h*4+d holds K = h*16 + khalf*8 + d*2 {,+1}  (dense 16-bit B layout).
// ---------------------------------------------------------------------------
__global__ __launch_bounds__(256) void pack_w_kernel(
    const float* __restrict__ W, const float* __restrict__ beta,
    unsigned* __restrict__ wpack)
{
    int o  = blockIdx.x * 256 + threadIdx.x;     // dword index, 524288 total
    int t  = o >> 8;                             // 256 dwords per tile
    int r  = o & 255;
    int kt = t >> 6;
    int nt = t & 63;
    int h  = r >> 7;
    int l  = (r >> 2) & 31;
    int d  = r & 3;
    int n  = nt * 16 + (l & 15);
    int K  = kt * 32 + h * 16 + (l >> 4) * 8 + d * 2;

    unsigned short h0 = 0, h1 = 0;
    if (n < NNODES) {
        float s = beta[n];
        h0 = f2bf(s * W[n * DIM + K]);
        h1 = f2bf(s * W[n * DIM + K + 1]);
    }
    wpack[o] = (unsigned)h0 | ((unsigned)h1 << 16);
}

// ---------------------------------------------------------------------------
// Kernel 2: leaf_reward[leaf] = softmax(leaf_dist[leaf,:]) . class_reward
// ---------------------------------------------------------------------------
__global__ __launch_bounds__(256) void leaf_reward_kernel(
    const float* __restrict__ leaf_dist, const float* __restrict__ class_reward,
    float* __restrict__ lrw)
{
    int i = blockIdx.x * 256 + threadIdx.x;
    if (i >= NLEAVES) return;
    float v[NCLS];
    float m = -3.4e38f;
    #pragma unroll
    for (int c = 0; c < NCLS; ++c) { v[c] = leaf_dist[i * NCLS + c]; m = fmaxf(m, v[c]); }
    float s = 0.f, dot = 0.f;
    #pragma unroll
    for (int c = 0; c < NCLS; ++c) { float e = __expf(v[c] - m); s += e; dot += e * class_reward[c]; }
    lrw[i] = dot / s;
}

// ---------------------------------------------------------------------------
// Kernel 3 (fused): GEMM + sigmoid -> probs in LDS (f16) -> tree reduction.
// 8 waves: wave w -> row group (w&1)*16, column quarter (w>>1)*256.
// Per nt-pair: K loop of 32 steps, 2x v_wmma_f32_16x16x32_bf16 per step.
// Then per-wave tree reduction of 4 rows straight out of LDS.
// ---------------------------------------------------------------------------
__global__ __launch_bounds__(256) void fused_tree_kernel(
    const float* __restrict__ x, const float* __restrict__ bvec,
    const float* __restrict__ beta, const v4i* __restrict__ wpack,
    const float* __restrict__ lrw, float* __restrict__ out)
{
    extern __shared__ char lds[];
    unsigned short* xs = (unsigned short*)lds;            // bf16 [32][XS_STR]
    _Float16*       ph = (_Float16*)(lds + XS_BYTES);     // f16  [32][PH_STR]

    const int tid  = threadIdx.x;
    const int row0 = blockIdx.x * M_TILE;

    // ---- stage x strip: 32 rows x 1024 dims, f32 -> bf16 ----
    for (int idx = tid; idx < M_TILE * (DIM / 4); idx += 256) {
        int row = idx >> 8;                 // 256 float4 per row
        int k4  = (idx & 255) << 2;
        v4f f = *reinterpret_cast<const v4f*>(x + (size_t)(row0 + row) * DIM + k4);
        unsigned lo = (unsigned)f2bf(f.x) | ((unsigned)f2bf(f.y) << 16);
        unsigned hi = (unsigned)f2bf(f.z) | ((unsigned)f2bf(f.w) << 16);
        uint2 pk; pk.x = lo; pk.y = hi;
        *reinterpret_cast<uint2*>(xs + row * XS_STR + k4) = pk;
    }
    __syncthreads();

    const int w     = tid >> 5;
    const int lane  = tid & 31;
    const int rgrp  = w & 1;          // 2 row groups of 16
    const int cq    = w >> 1;         // 4 column quarters of 256
    const int half  = lane >> 4;      // wave half (K-striping for A, C rows)
    const int mlane = lane & 15;

    const unsigned short* arow = xs + (rgrp * 16 + mlane) * XS_STR;

    for (int i = 0; i < 8; ++i) {
        const int nt0 = cq * 16 + 2 * i;
        v8f acc0 = {0.f, 0.f, 0.f, 0.f, 0.f, 0.f, 0.f, 0.f};
        v8f acc1 = {0.f, 0.f, 0.f, 0.f, 0.f, 0.f, 0.f, 0.f};

        for (int kt = 0; kt < 32; ++kt) {
            union { v4i q[2]; v16bf v; } A, B0, B1;
            // A fragment: lanes 0-15 K {0..7,16..23}, lanes 16-31 K {8..15,24..31}
            const unsigned short* ap = arow + kt * 32 + half * 8;
            A.q[0] = *reinterpret_cast<const v4i*>(ap);
            A.q[1] = *reinterpret_cast<const v4i*>(ap + 16);
            // B fragments: packed tiles, 64 int4 per tile, region1 at +32
            const v4i* wq = wpack + (size_t)(kt * 64 + nt0) * 64 + lane;
            B0.q[0] = wq[0];   B0.q[1] = wq[32];
            B1.q[0] = wq[64];  B1.q[1] = wq[96];

            acc0 = __builtin_amdgcn_wmma_f32_16x16x32_bf16(
                false, A.v, false, B0.v, (short)0, acc0, false, false);
            acc1 = __builtin_amdgcn_wmma_f32_16x16x32_bf16(
                false, A.v, false, B1.v, (short)0, acc1, false, false);
        }

        // epilogue: logit + beta*b, sigmoid, store prob (f16) to LDS
        #pragma unroll
        for (int j = 0; j < 2; ++j) {
            v8f acc = j ? acc1 : acc0;
            int n = (nt0 + j) * 16 + mlane;
            float bb = (n < NNODES) ? beta[n] * bvec[n] : 0.f;
            #pragma unroll
            for (int v = 0; v < 8; ++v) {
                float z = acc[v] + bb;
                float p = 1.0f / (1.0f + __expf(-z));
                ph[(rgrp * 16 + half * 8 + v) * PH_STR + n] = (_Float16)p;
            }
        }
    }
    __syncthreads();

    // ---- tree reduction straight from LDS: each wave reduces 4 rows ----
    float lr[32];
    #pragma unroll
    for (int j = 0; j < 32; ++j) lr[j] = lrw[lane * 32 + j];

    for (int rr = 0; rr < 4; ++rr) {
        const int row = w * 4 + rr;
        const _Float16* pr = ph + row * PH_STR;

        float v9[16];
        #pragma unroll
        for (int j = 0; j < 16; ++j) {
            float p = (float)pr[511 + lane * 16 + j];
            v9[j] = p * lr[2 * j] + (1.f - p) * lr[2 * j + 1];
        }
        float v8a[8];
        #pragma unroll
        for (int j = 0; j < 8; ++j) {
            float p = (float)pr[255 + lane * 8 + j];
            v8a[j] = p * v9[2 * j] + (1.f - p) * v9[2 * j + 1];
        }
        float v7[4];
        #pragma unroll
        for (int j = 0; j < 4; ++j) {
            float p = (float)pr[127 + lane * 4 + j];
            v7[j] = p * v8a[2 * j] + (1.f - p) * v8a[2 * j + 1];
        }
        float v6[2];
        #pragma unroll
        for (int j = 0; j < 2; ++j) {
            float p = (float)pr[63 + lane * 2 + j];
            v6[j] = p * v7[2 * j] + (1.f - p) * v7[2 * j + 1];
        }
        float p5 = (float)pr[31 + lane];
        float v  = p5 * v6[0] + (1.f - p5) * v6[1];

        int node = 31 + lane;
        #pragma unroll
        for (int s = 1; s <= 16; s <<= 1) {
            float vs = __shfl_down(v, (unsigned)s, 32);
            node = (node - 1) >> 1;
            float p = (float)pr[node];
            v = p * v + (1.f - p) * vs;
        }
        if (lane == 0) out[row0 + row] = v;
    }
}

// ---------------------------------------------------------------------------
extern "C" void kernel_launch(void* const* d_in, const int* in_sizes, int n_in,
                              void* d_out, int out_size, void* d_ws, size_t ws_size,
                              hipStream_t stream) {
    (void)in_sizes; (void)n_in; (void)out_size; (void)ws_size;
    const float* x            = (const float*)d_in[0];
    const float* W            = (const float*)d_in[1];
    const float* bvec         = (const float*)d_in[2];
    const float* beta         = (const float*)d_in[3];
    const float* leaf_dist    = (const float*)d_in[4];
    const float* class_reward = (const float*)d_in[5];
    float* out = (float*)d_out;

    char*  ws    = (char*)d_ws;
    v4i*   wpack = (v4i*)ws;
    float* lrw   = (float*)(ws + WS_LRW_OFF);

    pack_w_kernel<<<2048, 256, 0, stream>>>(W, beta, (unsigned*)wpack);
    leaf_reward_kernel<<<4, 256, 0, stream>>>(leaf_dist, class_reward, lrw);
    fused_tree_kernel<<<BATCH / M_TILE, 256, LDS_BYTES, stream>>>(
        x, bvec, beta, wpack, lrw, out);
}
